// LSTMDecoder_4784593568122
// MI455X (gfx1250) — matrix operand order repaired and checked by reading, not compiled, not run on hardware
//
#include <hip/hip_runtime.h>
#include <hip/hip_bf16.h>

// ---------------------------------------------------------------------------
// LSTM decoder w/ attention on gfx1250 (MI455X), all GEMMs via v_wmma_f32_16x16x32_f16.
//
//  * x @ W_ih.T hoisted out of the recurrence -> one big WMMA GEMM per layer.
//  * B == 16 == one WMMA M-tile: recurrent h @ W_hh.T is a wave-level WMMA job
//    (32 blocks x 4 waves, one wave per gate i/f/g/o).
//  * Attention softmax shift-invariance -> context is per-batch, computed once.
//  * Final FC (2048x1024)@(1024x32000) = 134 GFLOP dominates. v1 was L2-BW
//    bound (~12.8 FLOP/B). v2: block = 8 M-tiles x 64 N; weight (B) k-chunks
//    staged through double-buffered LDS and shared by all 8 waves ->
//    ~43 FLOP/B from L2, WMMA-bound.
// ---------------------------------------------------------------------------

typedef __attribute__((ext_vector_type(16))) _Float16 v16h;
typedef __attribute__((ext_vector_type(8)))  _Float16 v8h;
typedef __attribute__((ext_vector_type(8)))  float    v8f;

#define BB 16
#define SS 128
#define HH 512
#define EE 512
#define VV 32000
#define KK 196
#define G4H 2048   // 4*H

// ---- workspace layout (bytes) ----
static const size_t OFF_FCW16 = 0;                         // 32000*1024*2 = 65,536,000
static const size_t OFF_W0I16 = OFF_FCW16 + 65536000;      // 2048*512*2
static const size_t OFF_W0H16 = OFF_W0I16 + 2097152;
static const size_t OFF_W1I16 = OFF_W0H16 + 2097152;
static const size_t OFF_W1H16 = OFF_W1I16 + 2097152;
static const size_t OFF_X16   = OFF_W1H16 + 2097152;       // (S*B,512) f16
static const size_t OFF_XW    = OFF_X16   + 2097152;       // (S*B,2048) f32 (reused L0/L1)
static const size_t OFF_Y016  = OFF_XW    + 16777216;      // (S,B,512) f16
static const size_t OFF_Y116  = OFF_Y016  + 2097152;
static const size_t OFF_Y132  = OFF_Y116  + 2097152;       // (S,B,512) f32
static const size_t OFF_Z16   = OFF_Y132  + 4194304;       // (S*B,1024) f16 (also L0 f32 y dump)
static const size_t OFF_H016  = OFF_Z16   + 4194304;       // (B,512) f16
static const size_t OFF_CBUF  = OFF_H016  + 16384;         // (B,512) f32
static const size_t OFF_CTX   = OFF_CBUF  + 32768;         // (B,512) f32
static const size_t OFF_BIAS0 = OFF_CTX   + 32768;         // (2048,) f32
static const size_t OFF_BIAS1 = OFF_BIAS0 + 8192;

// ---------------------------------------------------------------------------
// WMMA fragment loaders, contiguous k order k = 16*(lane/16) + s (not the ISA table order; A and B match).
// A (16x32 f16, MxK): lanes 0-15 -> M=lane, K=0..15
//                     lanes 16-31 -> M=lane-16, K=16..31
__device__ __forceinline__ v16h load_a16(const _Float16* __restrict__ p, int ld, int kb) {
    int lane = threadIdx.x & 31;
    int m    = lane & 15;
    int klo  = (lane < 16) ? 0 : 16;
    const _Float16* r = p + (size_t)m * ld + kb + klo;
    return *(const v16h*)(r);
}
// B (32x16 f16, KxN) taken as W^T where W is (N,K) row-major -> column n of B is
// row n of W (contiguous!). Lanes 0-15: N=lane, K=0..15; lanes 16-31: K=16..31.
__device__ __forceinline__ v16h load_b16(const _Float16* __restrict__ W, int ldw, int n0, int kb) {
    int lane = threadIdx.x & 31;
    int n    = lane & 15;
    int kh   = (lane < 16) ? 0 : 16;
    return *(const v16h*)(W + (size_t)(n0 + n) * ldw + kb + kh);
}
// C/D f32 16x16: VGPR r -> M = r + (lane<16 ? 0 : 8), N = lane&15

__device__ __forceinline__ float sigmoidf_(float x) { return 1.0f / (1.0f + __expf(-x)); }

// ---------------------------------------------------------------------------
// small utility kernels
__global__ void k_f32_to_f16(_Float16* __restrict__ dst, const float* __restrict__ src, int n) {
    int i = blockIdx.x * blockDim.x + threadIdx.x;
    if (i < n) dst[i] = (_Float16)src[i];
}

__global__ void k_bias_sum(float* __restrict__ dst, const float* __restrict__ a,
                           const float* __restrict__ b, int n) {
    int i = blockIdx.x * blockDim.x + threadIdx.x;
    if (i < n) dst[i] = a[i] + b[i];
}

// embedding gather -> time-major f16 rows (row = s*16 + b)
__global__ void k_embed(_Float16* __restrict__ X, const float* __restrict__ emb,
                        const int* __restrict__ captions) {
    int idx = blockIdx.x * blockDim.x + threadIdx.x;       // over 2048*512
    if (idx >= SS * BB * EE) return;
    int r = idx >> 9, e = idx & 511;
    int s = r >> 4, b = r & 15;
    int tok = captions[b * (SS + 1) + s];                  // captions is (B, T=S+1)
    X[idx] = (_Float16)emb[(size_t)tok * EE + e];
}

__global__ void k_init_state(_Float16* __restrict__ h16, float* __restrict__ c,
                             const float* __restrict__ h0, const float* __restrict__ c0) {
    int i = blockIdx.x * blockDim.x + threadIdx.x;
    if (i >= BB * HH) return;
    h16[i] = (_Float16)h0[i];
    c[i]   = c0[i];
}

// ---------------------------------------------------------------------------
// Generic WMMA GEMM: C(M,N) = A(M,K) @ W(N,K)^T + bias.  1 wave = 16x64 tile.
// (Used for the hoisted x@W_ih.T GEMMs; ~4 GFLOP each, non-critical.)
__global__ void k_gemm_xw(const _Float16* __restrict__ A, const _Float16* __restrict__ W,
                          const float* __restrict__ bias, float* __restrict__ C,
                          int M, int N, int K) {
    int wid    = blockIdx.x * (blockDim.x >> 5) + (threadIdx.x >> 5);
    int mTiles = M >> 4;
    int mt = wid % mTiles;
    int ng = wid / mTiles;
    int n0 = ng * 64;
    if (n0 >= N) return;
    v8f acc[4] = {};
    const _Float16* Ab = A + (size_t)mt * 16 * K;
    for (int kb = 0; kb < K; kb += 32) {
        v16h a = load_a16(Ab, K, kb);
#pragma unroll
        for (int t = 0; t < 4; ++t) {
            v16h b = load_b16(W, K, n0 + t * 16, kb);
            acc[t] = __builtin_amdgcn_wmma_f32_16x16x32_f16(false, a, false, b,
                                                            (short)0, acc[t], false, false);
        }
    }
    int lane = threadIdx.x & 31;
    int mrow = (lane < 16) ? 0 : 8;
    int ncol = lane & 15;
#pragma unroll
    for (int t = 0; t < 4; ++t) {
#pragma unroll
        for (int r = 0; r < 8; ++r) {
            int mm = mt * 16 + mrow + r;
            int nn = n0 + t * 16 + ncol;
            C[(size_t)mm * N + nn] = acc[t][r] + bias[nn];
        }
    }
}

// ---------------------------------------------------------------------------
// One LSTM time step. grid = 32 (hidden 16-slices), block = 128 (4 waves = gates i,f,g,o).
// gates = hprev @ Whh^T (WMMA) + XW[s] (x@Wih^T + biases, precomputed).
__global__ void k_lstm_step(const _Float16* __restrict__ hprev,  // (16,512) f16
                            const _Float16* __restrict__ Whh,    // (2048,512) f16
                            const float* __restrict__ xw,        // (16,2048) f32 this step
                            float* __restrict__ c,               // (16,512) f32 in/out
                            _Float16* __restrict__ y16,          // (16,512) f16 out (next h)
                            float* __restrict__ y32) {           // (16,512) f32 out
    __shared__ float g_lds[4][16][16];
    int j    = blockIdx.x;          // hidden tile 0..31
    int gate = threadIdx.x >> 5;    // 0..3 = i,f,g,o
    int lane = threadIdx.x & 31;
    int n0   = gate * HH + j * 16;  // column base in (16,2048) gate matrix
    v8f acc = {};
    for (int kb = 0; kb < HH; kb += 32) {
        v16h a = load_a16(hprev, HH, kb);
        v16h b = load_b16(Whh, HH, n0, kb);
        acc = __builtin_amdgcn_wmma_f32_16x16x32_f16(false, a, false, b,
                                                     (short)0, acc, false, false);
    }
    int mrow = (lane < 16) ? 0 : 8;
    int ncol = lane & 15;
#pragma unroll
    for (int r = 0; r < 8; ++r) {
        int m = mrow + r;
        g_lds[gate][m][ncol] = acc[r] + xw[m * G4H + n0 + ncol];
    }
    __syncthreads();
    int e = threadIdx.x;
#pragma unroll
    for (int t = 0; t < 2; ++t, e += 128) {
        int m = e >> 4, n = e & 15;
        int h = j * 16 + n;
        float ig = sigmoidf_(g_lds[0][m][n]);
        float fg = sigmoidf_(g_lds[1][m][n]);
        float gg = tanhf(g_lds[2][m][n]);
        float og = sigmoidf_(g_lds[3][m][n]);
        float cn = fg * c[m * HH + h] + ig * gg;
        float hn = og * tanhf(cn);
        c[m * HH + h]   = cn;
        y32[m * HH + h] = hn;
        y16[m * HH + h] = (_Float16)hn;
    }
}

// ---------------------------------------------------------------------------
// Per-batch attention context. score = pimg[b,k] + f(b,s); f cancels in softmax
// over k, so attn (and ctx) are s-invariant -> compute once per batch element.
__global__ void k_attn_ctx(const float* __restrict__ img,     // (B,H,K)
                           const float* __restrict__ attn_w,  // first H entries used
                           float* __restrict__ ctx) {         // (B,H)
    __shared__ float sp[KK];
    __shared__ float red[256];
    int b = blockIdx.x;
    const float* ib = img + (size_t)b * HH * KK;
    for (int k = threadIdx.x; k < KK; k += blockDim.x) {
        float s = 0.f;
        for (int h = 0; h < HH; ++h) s += ib[h * KK + k] * attn_w[h];
        sp[k] = s;
    }
    __syncthreads();
    float m = -1e30f;
    for (int k = threadIdx.x; k < KK; k += blockDim.x) m = fmaxf(m, sp[k]);
    red[threadIdx.x] = m; __syncthreads();
    for (int st = 128; st > 0; st >>= 1) {
        if (threadIdx.x < st) red[threadIdx.x] = fmaxf(red[threadIdx.x], red[threadIdx.x + st]);
        __syncthreads();
    }
    m = red[0]; __syncthreads();
    float psum = 0.f;
    for (int k = threadIdx.x; k < KK; k += blockDim.x) {
        float e = __expf(sp[k] - m);
        sp[k] = e; psum += e;
    }
    red[threadIdx.x] = psum; __syncthreads();
    for (int st = 128; st > 0; st >>= 1) {
        if (threadIdx.x < st) red[threadIdx.x] += red[threadIdx.x + st];
        __syncthreads();
    }
    float inv = 1.0f / red[0];
    for (int h = threadIdx.x; h < HH; h += blockDim.x) {
        float a = 0.f;
        for (int k = 0; k < KK; ++k) a += sp[k] * ib[h * KK + k];
        ctx[b * HH + h] = a * inv;
    }
}

// Build FC input Z(2048,1024) f16 = [ masked y1 | broadcast ctx ]
__global__ void k_build_z(_Float16* __restrict__ Z, const float* __restrict__ y32,
                          const float* __restrict__ ctx, const int* __restrict__ seqlen) {
    int idx = blockIdx.x * blockDim.x + threadIdx.x;   // over 2048*1024
    if (idx >= SS * BB * 2 * HH) return;
    int r = idx >> 10, col = idx & 1023;
    int s = r >> 4, b = r & 15;
    float v;
    if (col < HH) {
        bool valid = s < (seqlen[b] - 1);
        v = valid ? y32[(size_t)r * HH + col] : 0.f;
    } else {
        v = ctx[b * HH + (col - HH)];
    }
    Z[idx] = (_Float16)v;
}

// ---------------------------------------------------------------------------
// Final FC v2: logits(b,s,:) = Z(row s*16+b) @ fc_w^T + fc_b.  134 GFLOP.
// Block = 256 thr (8 waves) computing 128 rows x 64 cols; wave w owns M-tile
// mg*8+w. Weight k-chunks (64x32 f16 = 4KB) double-buffered in LDS and shared
// by all 8 waves -> 8x less B traffic from L2 (~43 FLOP/B).
__global__ void k_fc(const _Float16* __restrict__ Z, const _Float16* __restrict__ W,
                     const float* __restrict__ bias, float* __restrict__ out) {
    __shared__ _Float16 sB[2][64 * 32];
    const int Kd = 2 * HH;                 // 1024
    int ng   = blockIdx.x >> 4;            // 0..499 -> n0 = ng*64
    int mg   = blockIdx.x & 15;            // 0..15  -> rows mg*128..+127
    int n0   = ng * 64;
    int wave = threadIdx.x >> 5;
    int lane = threadIdx.x & 31;
    int mt   = mg * 8 + wave;              // this wave's M-tile (== time step s)
    const _Float16* Ab = Z + (size_t)mt * 16 * Kd;

    // B staging: thread -> one 16B chunk of the 64x32 tile
    int srow = threadIdx.x >> 2;           // 0..63
    int scol = (threadIdx.x & 3) * 8;      // 0,8,16,24
    const _Float16* wsrc = W + (size_t)(n0 + srow) * Kd + scol;

    *(v8h*)&sB[0][srow * 32 + scol] = *(const v8h*)(wsrc);   // prefetch k-step 0

    v8f acc[4] = {};
    for (int kt = 0; kt < Kd / 32; ++kt) {
        __syncthreads();                   // staged data visible; prev reads done
        if (kt + 1 < Kd / 32) {            // stage next chunk into other buffer
            *(v8h*)&sB[(kt + 1) & 1][srow * 32 + scol] =
                *(const v8h*)(wsrc + (size_t)(kt + 1) * 32);
        }
        const _Float16* bb = &sB[kt & 1][0];
        v16h a = load_a16(Ab, Kd, kt * 32);
#pragma unroll
        for (int t = 0; t < 4; ++t) {
            v16h b = load_b16(bb, 32, t * 16, 0);
            acc[t] = __builtin_amdgcn_wmma_f32_16x16x32_f16(false, a, false, b,
                                                            (short)0, acc[t], false, false);
        }
    }
    int mrow = (lane < 16) ? 0 : 8;
    int ncol = lane & 15;
#pragma unroll
    for (int t = 0; t < 4; ++t) {
#pragma unroll
        for (int r = 0; r < 8; ++r) {
            int s = mt;                    // row = s*16 + batch
            int b = mrow + r;
            int nn = n0 + t * 16 + ncol;
            out[((size_t)b * SS + s) * VV + nn] = acc[t][r] + bias[nn];
        }
    }
}

// ---------------------------------------------------------------------------
extern "C" void kernel_launch(void* const* d_in, const int* in_sizes, int n_in,
                              void* d_out, int out_size, void* d_ws, size_t ws_size,
                              hipStream_t stream) {
    const int*   captions = (const int*)d_in[0];
    const float* img      = (const float*)d_in[1];
    const int*   seqlen   = (const int*)d_in[2];
    const float* h0       = (const float*)d_in[3];
    const float* c0       = (const float*)d_in[4];
    const float* emb      = (const float*)d_in[5];
    const float* W_ih0    = (const float*)d_in[6];
    const float* W_hh0    = (const float*)d_in[7];
    const float* b_ih0    = (const float*)d_in[8];
    const float* b_hh0    = (const float*)d_in[9];
    const float* W_ih1    = (const float*)d_in[10];
    const float* W_hh1    = (const float*)d_in[11];
    const float* b_ih1    = (const float*)d_in[12];
    const float* b_hh1    = (const float*)d_in[13];
    const float* attn_w   = (const float*)d_in[14];
    const float* fc_w     = (const float*)d_in[16];
    const float* fc_b     = (const float*)d_in[17];
    float* out = (float*)d_out;

    char* ws = (char*)d_ws;
    _Float16* FCW16 = (_Float16*)(ws + OFF_FCW16);
    _Float16* W0I16 = (_Float16*)(ws + OFF_W0I16);
    _Float16* W0H16 = (_Float16*)(ws + OFF_W0H16);
    _Float16* W1I16 = (_Float16*)(ws + OFF_W1I16);
    _Float16* W1H16 = (_Float16*)(ws + OFF_W1H16);
    _Float16* X16   = (_Float16*)(ws + OFF_X16);
    float*    XW    = (float*)   (ws + OFF_XW);
    _Float16* Y016  = (_Float16*)(ws + OFF_Y016);
    _Float16* Y116  = (_Float16*)(ws + OFF_Y116);
    float*    Y132  = (float*)   (ws + OFF_Y132);
    _Float16* Z16   = (_Float16*)(ws + OFF_Z16);
    float*    Y0DMP = (float*)   (ws + OFF_Z16);   // L0 f32 y is unused; dump into Z area
    _Float16* H016  = (_Float16*)(ws + OFF_H016);
    float*    CBUF  = (float*)   (ws + OFF_CBUF);
    float*    CTX   = (float*)   (ws + OFF_CTX);
    float*    BIAS0 = (float*)   (ws + OFF_BIAS0);
    float*    BIAS1 = (float*)   (ws + OFF_BIAS1);

    // --- precision conversions (weights read many times -> one-time f16 copies) ---
    k_f32_to_f16<<<(VV * 2 * HH + 255) / 256, 256, 0, stream>>>(FCW16, fc_w, VV * 2 * HH);
    k_f32_to_f16<<<(G4H * EE + 255) / 256, 256, 0, stream>>>(W0I16, W_ih0, G4H * EE);
    k_f32_to_f16<<<(G4H * HH + 255) / 256, 256, 0, stream>>>(W0H16, W_hh0, G4H * HH);
    k_f32_to_f16<<<(G4H * HH + 255) / 256, 256, 0, stream>>>(W1I16, W_ih1, G4H * HH);
    k_f32_to_f16<<<(G4H * HH + 255) / 256, 256, 0, stream>>>(W1H16, W_hh1, G4H * HH);
    k_bias_sum<<<(G4H + 255) / 256, 256, 0, stream>>>(BIAS0, b_ih0, b_hh0, G4H);
    k_bias_sum<<<(G4H + 255) / 256, 256, 0, stream>>>(BIAS1, b_ih1, b_hh1, G4H);

    // --- embedding gather (f16) ---
    k_embed<<<(SS * BB * EE + 255) / 256, 256, 0, stream>>>(X16, emb, captions);

    // --- layer 0: hoisted input GEMM, then 128 sequential WMMA steps ---
    k_init_state<<<(BB * HH + 255) / 256, 256, 0, stream>>>(H016, CBUF, h0, c0);
    k_gemm_xw<<<(SS * BB / 16) * (G4H / 64) / 8, 256, 0, stream>>>(
        X16, W0I16, BIAS0, XW, SS * BB, G4H, EE);
    for (int s = 0; s < SS; ++s) {
        const _Float16* hp = (s == 0) ? H016 : (Y016 + (size_t)(s - 1) * BB * HH);
        k_lstm_step<<<HH / 16, 128, 0, stream>>>(
            hp, W0H16, XW + (size_t)s * BB * G4H, CBUF,
            Y016 + (size_t)s * BB * HH, Y0DMP + (size_t)s * BB * HH);
    }

    // --- layer 1 ---
    k_init_state<<<(BB * HH + 255) / 256, 256, 0, stream>>>(H016, CBUF,
                                                            h0 + BB * HH, c0 + BB * HH);
    k_gemm_xw<<<(SS * BB / 16) * (G4H / 64) / 8, 256, 0, stream>>>(
        Y016, W1I16, BIAS1, XW, SS * BB, G4H, HH);
    for (int s = 0; s < SS; ++s) {
        const _Float16* hp = (s == 0) ? H016 : (Y116 + (size_t)(s - 1) * BB * HH);
        k_lstm_step<<<HH / 16, 128, 0, stream>>>(
            hp, W1H16, XW + (size_t)s * BB * G4H, CBUF,
            Y116 + (size_t)s * BB * HH, Y132 + (size_t)s * BB * HH);
    }

    // --- attention context (per-batch, softmax shift-invariance exploited) ---
    k_attn_ctx<<<BB, 256, 0, stream>>>(img, attn_w, CTX);

    // --- FC input assembly + 134 GFLOP WMMA projection (LDS-staged weights) ---
    k_build_z<<<(SS * BB * 2 * HH + 255) / 256, 256, 0, stream>>>(Z16, Y132, CTX, seqlen);
    k_fc<<<(VV / 64) * ((SS * BB / 16) / 8), 256, 0, stream>>>(Z16, FCW16, fc_b, out);
}
